// C2fDA_27831388078410
// MI455X (gfx1250) — matrix operand order repaired and checked by reading, not compiled, not run on hardware
//
#include <hip/hip_runtime.h>
#include <hip/hip_bf16.h>

typedef __attribute__((ext_vector_type(16))) __bf16 v16bf;
typedef __attribute__((ext_vector_type(8)))  __bf16 v8bf;
typedef __attribute__((ext_vector_type(8)))  float  v8f;

// Global-address-space views so fragment traffic lowers to global_load/store
typedef __attribute__((address_space(1))) const __bf16 gcbf16;
typedef __attribute__((address_space(1))) __bf16       gbf16;
typedef __attribute__((address_space(1))) const v8bf   gcv8bf;
typedef __attribute__((address_space(1))) const v16bf  gcv16bf;
typedef __attribute__((address_space(1))) v8bf         gv8bf;
typedef __attribute__((address_space(1))) v8f          gv8f;
typedef __attribute__((address_space(1))) float        gf32;

#define BB 8
#define HH 64
#define WWI 64
#define LL 4096

__device__ __forceinline__ float silu_f(float v) { return v / (1.0f + __expf(-v)); }

// A-fragment: 16x32 bf16, lane(0-15)=row m; elems 0-15: k=16*hh+e (16 contiguous)
__device__ __forceinline__ v16bf load_afrag(gcbf16* ap, int kb, int hh)
{
    v8bf lo = *(gcv8bf*)(ap + kb + 16 * hh);
    v8bf hi = *(gcv8bf*)(ap + kb + 16 * hh + 8);
    v16bf a;
#pragma unroll
    for (int i = 0; i < 8; ++i) { a[i] = lo[i]; a[8 + i] = hi[i]; }
    return a;
}

// B-fragment: 32x16 bf16, lane=col n, k = 16*hh + e (16 contiguous).
// Pointer is pre-clamped to a valid pixel: load unconditionally, then
// branchlessly zero OOB columns (v_cndmask, co-executes with WMMA).
__device__ __forceinline__ v16bf load_bfrag(gcbf16* bp, int kb, bool ok)
{
    v16bf b = *(gcv16bf*)(bp + kb);
    const v16bf z = {};
    return ok ? b : z;
}

__device__ __forceinline__ void wmma8(v8f acc[2][4], const v16bf a[2], const v16bf b[4])
{
#pragma unroll
    for (int mi = 0; mi < 2; ++mi)
#pragma unroll
        for (int ni = 0; ni < 4; ++ni)
            acc[mi][ni] = __builtin_amdgcn_wmma_f32_16x16x32_bf16(
                false, a[mi], false, b[ni], (short)0, acc[mi][ni], false, false);
}

// ---------------------------------------------------------------------------
// Generic bf16 WMMA GEMM (f32 accumulate), ping-pong double-buffered K loop.
//   C[m,n] = sum_t sum_k A[t, m, k] * Bsrc[pix(n,t), k]
// Wave tile: 32 (M) x 64 (N)  -> 2x4 WMMA 16x16 tiles, 8 v_wmma per K-step.
// REQUIRES K % 64 == 0 (true for all launches: 256/512/1280).
// A : tap-major weights, row-major [taps][Mtot][lda] (lda = K)
// Bm: "column-major" activations: element B[k,n] at Bm[n*ldb + k] (NHWC)
// taps==9: implicit im2col for 3x3/pad1 conv over a 64x64 image
// mode 0: bf16 NHWC store (out + n*ldc + m), optional bias+SiLU
// mode 1: f32 [n][m] store (out + n*ldc + m), optional bias
// mode 2: f32 NCHW store (out + m*ldc + n), optional SiLU
// ---------------------------------------------------------------------------
__global__ __launch_bounds__(128)
void gemm_bf16_wmma(const __bf16* __restrict__ A,
                    const __bf16* __restrict__ Bm,
                    void* __restrict__ Cv,
                    const float* __restrict__ bias,
                    int Mtot, int K, int lda, int ldb, int ldc,
                    long long strideB, long long strideC,
                    int taps, int silu, int mode)
{
    const int lane  = threadIdx.x & 31;
    const int wave  = threadIdx.x >> 5;
    const int batch = blockIdx.z;
    const int n0    = blockIdx.x * 256 + wave * 64;   // 64 output columns (pixels)
    const int m0    = blockIdx.y * 32;                // 32 output rows (channels)

    const int hh  = lane >> 4;    // lane half (0/1)
    const int l15 = lane & 15;

    const __bf16* Bb = Bm + (long long)batch * strideB;

    const int y  = n0 >> 6;       // image row (n0 is a multiple of 64)
    const int xb = n0 & 63;

    v8f acc[2][4] = {};

    for (int t = 0; t < taps; ++t) {
        gcbf16* bp[4];
        bool ok[4];
        if (taps == 9) {
            const int dx = t % 3 - 1;
            const int ys = t / 3 - 1 + y;
            const bool rok = (unsigned)ys < (unsigned)HH;
            const int yc = min(max(ys, 0), HH - 1);
#pragma unroll
            for (int j = 0; j < 4; ++j) {
                const int xj = xb + 16 * j + l15 + dx;
                ok[j] = rok && ((unsigned)xj < (unsigned)WWI);
                const int xc = min(max(xj, 0), WWI - 1);
                bp[j] = (gcbf16*)(Bb + ((long long)yc * WWI + xc) * ldb + 16 * hh);
            }
        } else {
#pragma unroll
            for (int j = 0; j < 4; ++j) {
                ok[j] = true;
                bp[j] = (gcbf16*)(Bb + (long long)(n0 + 16 * j + l15) * ldb + 16 * hh);
            }
        }
        const __bf16* ap0g = A + ((long long)t * Mtot + (m0 + l15)) * lda;
        gcbf16* ap0 = (gcbf16*)ap0g;
        gcbf16* ap1 = (gcbf16*)(ap0g + 16LL * lda);

        // ---- prologue: stage-0 fragments (kb = 0) ----
        v16bf a0[2], b0[4], a1[2], b1[4];
        a0[0] = load_afrag(ap0, 0, hh);
        a0[1] = load_afrag(ap1, 0, hh);
#pragma unroll
        for (int j = 0; j < 4; ++j) b0[j] = load_bfrag(bp[j], 0, ok[j]);

        int kb = 0;
        for (; kb + 64 < K; kb += 64) {
            // stage-1 loads (kb+32) issued before stage-0 math
            a1[0] = load_afrag(ap0, kb + 32, hh);
            a1[1] = load_afrag(ap1, kb + 32, hh);
#pragma unroll
            for (int j = 0; j < 4; ++j) b1[j] = load_bfrag(bp[j], kb + 32, ok[j]);
            __builtin_prefetch(ap0g + kb + 96, 0, 1);

            wmma8(acc, a0, b0);       // math on kb

            // stage-0 loads (kb+64) issued before stage-1 math
            a0[0] = load_afrag(ap0, kb + 64, hh);
            a0[1] = load_afrag(ap1, kb + 64, hh);
#pragma unroll
            for (int j = 0; j < 4; ++j) b0[j] = load_bfrag(bp[j], kb + 64, ok[j]);

            wmma8(acc, a1, b1);       // math on kb+32
        }
        // ---- epilogue: last 64 of K (kb == K-64) ----
        a1[0] = load_afrag(ap0, kb + 32, hh);
        a1[1] = load_afrag(ap1, kb + 32, hh);
#pragma unroll
        for (int j = 0; j < 4; ++j) b1[j] = load_bfrag(bp[j], kb + 32, ok[j]);
        wmma8(acc, a0, b0);
        wmma8(acc, a1, b1);
    }

    // ---- store: lane(0-15)=col n, VGPR g -> row m = m0 + 16*mt + 8*hh + g ----
#pragma unroll
    for (int mt = 0; mt < 2; ++mt)
#pragma unroll
        for (int nt = 0; nt < 4; ++nt) {
            const int n  = n0 + nt * 16 + l15;
            const int mb = m0 + mt * 16 + 8 * hh;
            if (mode == 0) {
                __bf16* Co = (__bf16*)Cv + (long long)batch * strideC +
                             (long long)n * ldc + mb;
                v8bf ov;
#pragma unroll
                for (int g = 0; g < 8; ++g) {
                    float v = acc[mt][nt][g];
                    if (bias) v += bias[mb + g];
                    if (silu) v = silu_f(v);
                    ov[g] = (__bf16)v;
                }
                *(gv8bf*)(gbf16*)Co = ov;
            } else if (mode == 1) {
                float* Co = (float*)Cv + (long long)batch * strideC +
                            (long long)n * ldc + mb;
                v8f ov;
#pragma unroll
                for (int g = 0; g < 8; ++g) {
                    float v = acc[mt][nt][g];
                    if (bias) v += bias[mb + g];
                    ov[g] = v;
                }
                *(gv8f*)(gf32*)Co = ov;
            } else {
                gf32* Co = (gf32*)((float*)Cv + (long long)batch * strideC);
#pragma unroll
                for (int g = 0; g < 8; ++g) {
                    float v = acc[mt][nt][g];
                    if (silu) v = silu_f(v);
                    Co[(long long)(mb + g) * ldc + n] = v;
                }
            }
        }
}

// ---------------------------------------------------------------------------
// Deformable-attention sampling: one wave per (b, l, head); lane = channel d.
// value: f32 [B][L][256]  off: f32 [B][L][64]  awl: f32 [B][L][32]
// bbox:  f32 [B][L][1][2] samp(out): bf16 [B][L][256]
// ---------------------------------------------------------------------------
__global__ __launch_bounds__(256)
void msdeform_sample(const float* __restrict__ value,
                     const float* __restrict__ offs,
                     const float* __restrict__ awl,
                     const float* __restrict__ bbox,
                     const int* __restrict__ vshapes,
                     __bf16* __restrict__ samp)
{
    const int lane = threadIdx.x & 31;
    const int h    = threadIdx.x >> 5;             // head 0..7
    const long long bl = blockIdx.x;               // b*L + l
    const long long b  = bl >> 12;                 // L = 4096

    const int Hi = vshapes[0], Wi = vshapes[1];
    const float Hf = (float)Hi, Wf = (float)Wi;

    const float* op = offs + bl * 64 + h * 8;
    const float* ap = awl + bl * 32 + h * 4;
    const float a0 = ap[0], a1 = ap[1], a2 = ap[2], a3 = ap[3];
    const float mx = fmaxf(fmaxf(a0, a1), fmaxf(a2, a3));
    const float e0 = __expf(a0 - mx), e1 = __expf(a1 - mx),
                e2 = __expf(a2 - mx), e3 = __expf(a3 - mx);
    const float inv = 1.0f / (e0 + e1 + e2 + e3);
    const float wp[4] = {e0 * inv, e1 * inv, e2 * inv, e3 * inv};

    const float bx = bbox[bl * 2 + 0];
    const float by = bbox[bl * 2 + 1];

    const float* vb = value + ((b << 12) * 256) + h * 32 + lane;

    float acc = 0.0f;
#pragma unroll
    for (int p = 0; p < 4; ++p) {
        const float gx = (bx + op[2 * p]     / Wf) * Wf - 0.5f;
        const float gy = (by + op[2 * p + 1] / Hf) * Hf - 0.5f;
        const float fx = floorf(gx), fy = floorf(gy);
        const float wx1 = gx - fx, wy1 = gy - fy;
        const float wx0 = 1.0f - wx1, wy0 = 1.0f - wy1;
        const int ix = (int)fx, iy = (int)fy;

        float s00 = 0.f, s10 = 0.f, s01 = 0.f, s11 = 0.f;
        const bool vx0 = (unsigned)ix < (unsigned)Wi;
        const bool vx1 = (unsigned)(ix + 1) < (unsigned)Wi;
        const bool vy0 = (unsigned)iy < (unsigned)Hi;
        const bool vy1 = (unsigned)(iy + 1) < (unsigned)Hi;
        if (vx0 && vy0) s00 = vb[(long long)(iy * Wi + ix) * 256];
        if (vx1 && vy0) s10 = vb[(long long)(iy * Wi + ix + 1) * 256];
        if (vx0 && vy1) s01 = vb[(long long)((iy + 1) * Wi + ix) * 256];
        if (vx1 && vy1) s11 = vb[(long long)((iy + 1) * Wi + ix + 1) * 256];

        acc += wp[p] * (s00 * wx0 * wy0 + s10 * wx1 * wy0 +
                        s01 * wx0 * wy1 + s11 * wx1 * wy1);
    }
    samp[bl * 256 + h * 32 + lane] = (__bf16)acc;
}

// ---------------------------------------------------------------------------
// Conversion kernels (f32 -> bf16 with layout changes)
// ---------------------------------------------------------------------------
__global__ void cvt_copy_bf16(const float* __restrict__ s, __bf16* __restrict__ d,
                              long long n)
{
    long long i = (long long)blockIdx.x * blockDim.x + threadIdx.x;
    if (i < n) d[i] = (__bf16)s[i];
}

__global__ void cvt_transpose_bf16(const float* __restrict__ s,
                                   __bf16* __restrict__ d, int rows, int cols)
{
    // s[r][c] (rows x cols) -> d[c][r]
    long long i = (long long)blockIdx.x * blockDim.x + threadIdx.x;
    if (i < (long long)rows * cols) {
        int c = (int)(i % cols);
        int r = (int)(i / cols);
        d[(long long)c * rows + r] = (__bf16)s[i];
    }
}

__global__ void cvt_tap_bf16(const float* __restrict__ s, __bf16* __restrict__ d,
                             int OC, int IC)
{
    // OIHW [oc][ic][3][3] -> tap-major [t][oc][ic]
    long long i = (long long)blockIdx.x * blockDim.x + threadIdx.x;
    if (i < (long long)OC * IC * 9) {
        int t = (int)(i % 9);
        long long r = i / 9;
        int ic = (int)(r % IC);
        int oc = (int)(r / IC);
        d[((long long)t * OC + oc) * IC + ic] = (__bf16)s[i];
    }
}

__global__ void cvt_nchw_nhwc_bf16(const float* __restrict__ s,
                                   __bf16* __restrict__ d, int Cc)
{
    long long i = (long long)blockIdx.x * blockDim.x + threadIdx.x;
    if (i < (long long)BB * Cc * LL) {
        int l = (int)(i % LL);
        long long r = i / LL;
        int c = (int)(r % Cc);
        int b = (int)(r / Cc);
        d[((long long)b * LL + l) * Cc + c] = (__bf16)s[i];
    }
}

// ---------------------------------------------------------------------------
extern "C" void kernel_launch(void* const* d_in, const int* in_sizes, int n_in,
                              void* d_out, int out_size, void* d_ws, size_t ws_size,
                              hipStream_t stream)
{
    (void)in_sizes; (void)n_in; (void)out_size; (void)ws_size;

    const float* x       = (const float*)d_in[0];
    const float* bbox    = (const float*)d_in[1];
    const int*   vshapes = (const int*)d_in[2];
    const float* cv1_w   = (const float*)d_in[3];
    const float* m0a_w   = (const float*)d_in[4];
    const float* m0b_w   = (const float*)d_in[5];
    const float* m1a_w   = (const float*)d_in[6];
    const float* m1b_w   = (const float*)d_in[7];
    const float* vproj_w = (const float*)d_in[8];
    const float* vproj_b = (const float*)d_in[9];
    const float* off_w   = (const float*)d_in[10];
    const float* off_b   = (const float*)d_in[11];
    const float* aw_w    = (const float*)d_in[12];
    const float* aw_b    = (const float*)d_in[13];
    const float* out_w   = (const float*)d_in[14];
    const float* out_b   = (const float*)d_in[15];
    const float* cv2_w   = (const float*)d_in[16];

    char* ws = (char*)d_ws;
    size_t pos = 0;
    auto alloc = [&](size_t bytes) -> void* {
        void* p = ws + pos;
        pos += (bytes + 255) & ~(size_t)255;
        return p;
    };

    __bf16* wx    = (__bf16*)alloc((size_t)BB * LL * 512 * 2);   // x, NHWC bf16
    __bf16* cat   = (__bf16*)alloc((size_t)BB * LL * 1280 * 2);  // [a|b|b1|b2|attn]
    __bf16* tbuf  = (__bf16*)alloc((size_t)BB * LL * 256 * 2);   // conv intermediate
    float*  val   = (float*) alloc((size_t)BB * LL * 256 * 4);   // value (f32)
    float*  offb  = (float*) alloc((size_t)BB * LL * 64 * 4);    // offsets (f32)
    float*  awb   = (float*) alloc((size_t)BB * LL * 32 * 4);    // aw logits (f32)
    __bf16* samp  = (__bf16*)alloc((size_t)BB * LL * 256 * 2);   // sampled (bf16)
    __bf16* wcv1  = (__bf16*)alloc((size_t)512 * 512 * 2);
    __bf16* wm0a  = (__bf16*)alloc((size_t)9 * 256 * 256 * 2);
    __bf16* wm0b  = (__bf16*)alloc((size_t)9 * 256 * 256 * 2);
    __bf16* wm1a  = (__bf16*)alloc((size_t)9 * 256 * 256 * 2);
    __bf16* wm1b  = (__bf16*)alloc((size_t)9 * 256 * 256 * 2);
    __bf16* wvpr  = (__bf16*)alloc((size_t)256 * 256 * 2);
    __bf16* woff  = (__bf16*)alloc((size_t)64 * 256 * 2);
    __bf16* waw   = (__bf16*)alloc((size_t)32 * 256 * 2);
    __bf16* wout  = (__bf16*)alloc((size_t)256 * 256 * 2);
    __bf16* wcv2  = (__bf16*)alloc((size_t)512 * 1280 * 2);

    auto nb = [](long long n) { return (unsigned)((n + 255) / 256); };

    // --- weight / input conversions ---
    cvt_copy_bf16<<<nb(512 * 512), 256, 0, stream>>>(cv1_w, wcv1, 512 * 512);
    cvt_copy_bf16<<<nb(512 * 1280), 256, 0, stream>>>(cv2_w, wcv2, 512 * 1280);
    cvt_tap_bf16<<<nb(9LL * 256 * 256), 256, 0, stream>>>(m0a_w, wm0a, 256, 256);
    cvt_tap_bf16<<<nb(9LL * 256 * 256), 256, 0, stream>>>(m0b_w, wm0b, 256, 256);
    cvt_tap_bf16<<<nb(9LL * 256 * 256), 256, 0, stream>>>(m1a_w, wm1a, 256, 256);
    cvt_tap_bf16<<<nb(9LL * 256 * 256), 256, 0, stream>>>(m1b_w, wm1b, 256, 256);
    cvt_transpose_bf16<<<nb(256 * 256), 256, 0, stream>>>(vproj_w, wvpr, 256, 256);
    cvt_transpose_bf16<<<nb(256 * 64), 256, 0, stream>>>(off_w, woff, 256, 64);
    cvt_transpose_bf16<<<nb(256 * 32), 256, 0, stream>>>(aw_w, waw, 256, 32);
    cvt_transpose_bf16<<<nb(256 * 256), 256, 0, stream>>>(out_w, wout, 256, 256);
    cvt_nchw_nhwc_bf16<<<nb(8LL * 512 * LL), 256, 0, stream>>>(x, wx, 512);

    const long long sc = (long long)LL * 1280;   // cat batch stride (elements)
    const dim3 blk(128);
    auto grd = [](int M) { return dim3(LL / 256, M / 32, BB); };

    // cv1: x(512) -> cat[0:512), SiLU
    gemm_bf16_wmma<<<grd(512), blk, 0, stream>>>(wcv1, wx, cat, nullptr,
        512, 512, 512, 512, 1280, (long long)LL * 512, sc, 1, 1, 0);
    // m0_cv1: b = cat[256:512) -> tbuf, 3x3, SiLU
    gemm_bf16_wmma<<<grd(256), blk, 0, stream>>>(wm0a, cat + 256, tbuf, nullptr,
        256, 256, 256, 1280, 256, sc, (long long)LL * 256, 9, 1, 0);
    // m0_cv2: tbuf -> cat[512:768) = b1, 3x3, SiLU
    gemm_bf16_wmma<<<grd(256), blk, 0, stream>>>(wm0b, tbuf, cat + 512, nullptr,
        256, 256, 256, 256, 1280, (long long)LL * 256, sc, 9, 1, 0);
    // m1_cv1: b1 -> tbuf, 3x3, SiLU
    gemm_bf16_wmma<<<grd(256), blk, 0, stream>>>(wm1a, cat + 512, tbuf, nullptr,
        256, 256, 256, 1280, 256, sc, (long long)LL * 256, 9, 1, 0);
    // m1_cv2: tbuf -> cat[768:1024) = b2 = q, 3x3, SiLU
    gemm_bf16_wmma<<<grd(256), blk, 0, stream>>>(wm1b, tbuf, cat + 768, nullptr,
        256, 256, 256, 256, 1280, (long long)LL * 256, sc, 9, 1, 0);
    // value = q @ vproj + b  (f32 [l][256])
    gemm_bf16_wmma<<<grd(256), blk, 0, stream>>>(wvpr, cat + 768, val, vproj_b,
        256, 256, 256, 1280, 256, sc, (long long)LL * 256, 1, 0, 1);
    // off = q @ off_w + b    (f32 [l][64])
    gemm_bf16_wmma<<<grd(64), blk, 0, stream>>>(woff, cat + 768, offb, off_b,
        64, 256, 256, 1280, 64, sc, (long long)LL * 64, 1, 0, 1);
    // aw logits = q @ aw_w + b (f32 [l][32]; softmax in sampler)
    gemm_bf16_wmma<<<grd(32), blk, 0, stream>>>(waw, cat + 768, awb, aw_b,
        32, 256, 256, 1280, 32, sc, (long long)LL * 32, 1, 0, 1);
    // bilinear sampling + softmax + point-weighted sum
    msdeform_sample<<<BB * LL, 256, 0, stream>>>(val, offb, awb, bbox, vshapes, samp);
    // attn = samp @ out_w + out_b -> cat[1024:1280), no activation
    gemm_bf16_wmma<<<grd(256), blk, 0, stream>>>(wout, samp, cat + 1024, out_b,
        256, 256, 256, 256, 1280, (long long)LL * 256, sc, 1, 0, 0);
    // cv2: cat(1280) -> d_out f32 NCHW, SiLU
    gemm_bf16_wmma<<<grd(512), blk, 0, stream>>>(wcv2, cat, (float*)d_out, nullptr,
        512, 1280, 1280, 1280, LL, sc, (long long)512 * LL, 1, 1, 2);
}
